// spk_vq_vae_resnet_50294067036780
// MI455X (gfx1250) — matrix operand order repaired and checked by reading, not compiled, not run on hardware
//
#include <hip/hip_runtime.h>

// ---------------------------------------------------------------------------
// CDNA5 (gfx1250) VQ-VAE forward.  All conv / convT / VQ-distance GEMMs run
// through v_wmma_f32_16x16x32_bf16 (bf16 A/B, f32 accumulate), wave32.
// Conv = tap-decomposed implicit GEMM: per kernel-tap k it is a clean GEMM
// over Cin (K dim), addressing is base + ci*stride (no div/mod), predication
// is value-select (cndmask).  Each wave: 16(M) x 32(N) output tile.
// VQ argmin: WMMA distance tiles + fully branch-free select-based tracking.
// ---------------------------------------------------------------------------

typedef __attribute__((ext_vector_type(16))) __bf16 v16bf;
typedef __attribute__((ext_vector_type(8)))  float  v8f;

// wmode 0: W(co,ci,k) = w[co][ci][k]            (conv,   w [Cout,Cin,K])
// wmode 1: W(co,ci,k) = w[ci][co][K-1-k]        (convT s=1, w [Cin,Cout,K])
// wmode 2: W(co,ci)   = w[ci][co][wtap], 1 tap  (convT s=2 k=2 parity pass)
template <int WMODE>
__global__ __launch_bounds__(32) void conv1d_wmma_kernel(
    const float* __restrict__ x, const float* __restrict__ w,
    const float* __restrict__ bias, float* __restrict__ y,
    int Bn, int Cin, int Lin, int Cout, int Nout, int Lout,
    int Ksz, int pad, int in_step, int out_step, int out_off, int wtap)
{
  const int lane = (int)threadIdx.x;       // wave32
  const int hl   = lane >> 4;              // lane half
  const int col  = lane & 15;
  const int nt   = (int)blockIdx.x * 32;   // N tile (B*Nout), 32 wide
  const int mt   = (int)blockIdx.y * 16;   // M tile (Cout)
  const int Ntot = Bn * Nout;

  const int  ma     = mt + col;            // A row = out channel
  const bool mvalid = (ma < Cout);
  const int  mac    = mvalid ? ma : 0;

  const int  nb0 = nt + col,       nb1 = nt + 16 + col;
  const bool nv0 = (nb0 < Ntot),   nv1 = (nb1 < Ntot);
  const int  nc0 = nv0 ? nb0 : 0,  nc1 = nv1 ? nb1 : 0;
  const int  bb0 = nc0 / Nout,     bb1 = nc1 / Nout;     // once per kernel
  const int  nn0 = nc0 - bb0*Nout, nn1 = nc1 - bb1*Nout;

  v8f acc0 = {0.f,0.f,0.f,0.f,0.f,0.f,0.f,0.f};
  v8f acc1 = {0.f,0.f,0.f,0.f,0.f,0.f,0.f,0.f};

  const int Ktap  = (WMODE == 2) ? 1 : Ksz;
  const int cinm1 = Cin - 1;

  for (int k = 0; k < Ktap; ++k) {
    // A base/stride for this tap (all 32-bit, no div/mod)
    const float* wp; int wstr;
    if (WMODE == 0)      { wp = w + mac * (Cin * Ksz) + k;     wstr = Ksz; }
    else if (WMODE == 1) { wp = w + mac * Ksz + (Ksz - 1 - k); wstr = Cout * Ksz; }
    else                 { wp = w + mac * 2 + wtap;            wstr = Cout * 2; }

    const int kk  = (WMODE == 2) ? 0 : k;
    const int ip0 = nn0 * in_step + kk - pad;
    const int ip1 = nn1 * in_step + kk - pad;
    const bool ib0 = nv0 && (ip0 >= 0) && (ip0 < Lin);
    const bool ib1 = nv1 && (ip1 >= 0) && (ip1 < Lin);
    const float* xp0 = x + (bb0 * Cin) * Lin + (ib0 ? ip0 : 0);
    const float* xp1 = x + (bb1 * Cin) * Lin + (ib1 ? ip1 : 0);

    for (int c0 = 0; c0 < Cin; c0 += 32) {
      v16bf a, b0, b1;
#pragma unroll
      for (int e = 0; e < 16; ++e) {
        // A: K 0..15 / 16..31 by lane half (matches B)
        int  cA  = c0 + hl * 16 + e;
        int  cAc = (cA > cinm1) ? cinm1 : cA;          // clamp, keep load legal
        float av = wp[cAc * wstr];
        a[e] = (__bf16)((mvalid && cA <= cinm1) ? av : 0.f);
        // B: K 0..15 / 16..31 by lane half (matches A)
        int  cB  = c0 + hl * 16 + e;
        int  cBc = (cB > cinm1) ? cinm1 : cB;
        bool cok = (cB <= cinm1);
        float bv0 = xp0[cBc * Lin];
        float bv1 = xp1[cBc * Lin];
        b0[e] = (__bf16)((ib0 && cok) ? bv0 : 0.f);
        b1[e] = (__bf16)((ib1 && cok) ? bv1 : 0.f);
      }
      acc0 = __builtin_amdgcn_wmma_f32_16x16x32_bf16(false, a, false, b0,
                                                     (short)0, acc0, false, false);
      acc1 = __builtin_amdgcn_wmma_f32_16x16x32_bf16(false, a, false, b1,
                                                     (short)0, acc1, false, false);
    }
  }

  const int o0 = nn0 * out_step + out_off;
  const int o1 = nn1 * out_step + out_off;
#pragma unroll
  for (int r = 0; r < 8; ++r) {
    int mg = mt + r + hl * 8;                 // D layout: M = r + 8*laneHalf
    if (mg < Cout) {
      float bsv = bias[mg];
      if (nv0) y[(bb0 * Cout + mg) * Lout + o0] = acc0[r] + bsv;
      if (nv1) y[(bb1 * Cout + mg) * Lout + o1] = acc1[r] + bsv;
    }
  }
}

// ---- BatchNorm (training mode, batch stats over (B,L)) --------------------
__global__ void bn_stats_kernel(const float* __restrict__ y, float* __restrict__ sums,
                                int Bn, int C, int L) {
  const int c = blockIdx.x, t = threadIdx.x;
  float s = 0.f, q = 0.f;
  for (int b = 0; b < Bn; ++b) {
    const float* row = y + (b * C + c) * L;
    for (int l = t; l < L; l += 256) { float v = row[l]; s += v; q += v * v; }
  }
  __shared__ float sh[256], sq[256];
  sh[t] = s; sq[t] = q;
  __syncthreads();
  for (int o = 128; o > 0; o >>= 1) {
    if (t < o) { sh[t] += sh[t + o]; sq[t] += sq[t + o]; }
    __syncthreads();
  }
  if (t == 0) { sums[2 * c] = sh[0]; sums[2 * c + 1] = sq[0]; }
}

__global__ void bn_finalize_kernel(const float* __restrict__ sums,
                                   const float* __restrict__ g,
                                   const float* __restrict__ beta,
                                   float* __restrict__ scsh, int C, float inv_cnt) {
  int c = blockIdx.x * blockDim.x + threadIdx.x;
  if (c < C) {
    float m = sums[2 * c] * inv_cnt;
    float v = sums[2 * c + 1] * inv_cnt - m * m;       // biased var (jnp default)
    float sc = g[c] * rsqrtf(v + 1e-5f);
    scsh[2 * c]     = sc;
    scsh[2 * c + 1] = beta[c] - m * sc;
  }
}

// one block per (b,c) row; only rowid % C needed
__global__ void bn_apply_kernel(const float* __restrict__ y, const float* __restrict__ scsh,
                                const float* __restrict__ addsrc, float* __restrict__ out,
                                int relu, int C, int L) {
  const int rowid = blockIdx.x;            // b*C + c
  const int c = rowid % C;
  const float sc = scsh[2 * c], sh = scsh[2 * c + 1];
  const int base = rowid * L;
  for (int l = threadIdx.x; l < L; l += blockDim.x) {
    float v = y[base + l] * sc + sh;
    if (addsrc) v += addsrc[base + l];
    if (relu)   v = fmaxf(v, 0.f);
    out[base + l] = v;
  }
}

// ---- Vector quantization --------------------------------------------------
__global__ void wnorm_kernel(const float* __restrict__ W, float* __restrict__ wn,
                             int Kc, int D) {
  int j = blockIdx.x * blockDim.x + threadIdx.x;
  if (j < Kc) {
    const float4* wr = (const float4*)(W + j * D);
    float s = 0.f;
    for (int q = 0; q < D / 4; ++q) {
      float4 v = wr[q];
      s += v.x * v.x + v.y * v.y + v.z * v.z + v.w * v.w;
    }
    wn[j] = s;
  }
}

// dist(n,j) = ||W_j||^2 - 2 <z_n, W_j>   (||z||^2 is argmin-invariant)
__global__ __launch_bounds__(32) void vq_argmin_kernel(
    const float* __restrict__ z, const float* __restrict__ W,
    const float* __restrict__ wn, int* __restrict__ idx, int Nrows, int Ncodes)
{
  const int lane = (int)threadIdx.x;
  const int hl   = lane >> 4, klow = hl * 8;
  const int col  = lane & 15;
  const int row  = (int)blockIdx.x * 16 + col;
  const int rc   = (row < Nrows) ? row : 0;

  // z fragment: B layout => 16 contiguous floats at z + row*32 + hl*16
  const float4* zp = (const float4*)(z + rc * 32 + hl * 16);
  float4 z0 = zp[0], z1 = zp[1], z2 = zp[2], z3 = zp[3];
  v16bf bz;
  bz[0]=(__bf16)z0.x;  bz[1]=(__bf16)z0.y;  bz[2]=(__bf16)z0.z;  bz[3]=(__bf16)z0.w;
  bz[4]=(__bf16)z1.x;  bz[5]=(__bf16)z1.y;  bz[6]=(__bf16)z1.z;  bz[7]=(__bf16)z1.w;
  bz[8]=(__bf16)z2.x;  bz[9]=(__bf16)z2.y;  bz[10]=(__bf16)z2.z; bz[11]=(__bf16)z2.w;
  bz[12]=(__bf16)z3.x; bz[13]=(__bf16)z3.y; bz[14]=(__bf16)z3.z; bz[15]=(__bf16)z3.w;

  float best = 3.4e38f; int bi = 0;
  for (int mtile = 0; mtile < Ncodes; mtile += 16) {
    // A fragment: rows = codes; K runs [hl*16, hl*16+16) (matches B)
    const float* wr = W + (mtile + col) * 32;
    const float4* wa = (const float4*)(wr + hl * 16);
    const float4* wb = (const float4*)(wr + hl * 16 + 8);
    float4 a0 = wa[0], a1 = wa[1], a2 = wb[0], a3 = wb[1];
    v16bf a;
    a[0]=(__bf16)a0.x;  a[1]=(__bf16)a0.y;  a[2]=(__bf16)a0.z;  a[3]=(__bf16)a0.w;
    a[4]=(__bf16)a1.x;  a[5]=(__bf16)a1.y;  a[6]=(__bf16)a1.z;  a[7]=(__bf16)a1.w;
    a[8]=(__bf16)a2.x;  a[9]=(__bf16)a2.y;  a[10]=(__bf16)a2.z; a[11]=(__bf16)a2.w;
    a[12]=(__bf16)a3.x; a[13]=(__bf16)a3.y; a[14]=(__bf16)a3.z; a[15]=(__bf16)a3.w;

    v8f acc = {0.f,0.f,0.f,0.f,0.f,0.f,0.f,0.f};
    acc = __builtin_amdgcn_wmma_f32_16x16x32_bf16(false, a, false, bz,
                                                  (short)0, acc, false, false);

    // ||W||^2 for this lane's 8 M-rows: two contiguous float4 loads
    const float4* wq = (const float4*)(wn + mtile + klow);
    float4 n0 = wq[0], n1 = wq[1];
    float wnv[8] = {n0.x, n0.y, n0.z, n0.w, n1.x, n1.y, n1.z, n1.w};

    // branch-free min tracking; mg scanned ascending -> strict '<' keeps
    // the lowest index (jnp.argmin semantics) within this lane
#pragma unroll
    for (int r = 0; r < 8; ++r) {
      float d  = wnv[r] - 2.f * acc[r];
      int   mg = mtile + klow + r;
      bool take = d < best;
      best = take ? d : best;
      bi   = take ? mg : bi;
    }
  }
  // merge the two lane halves (M 0..7 vs 8..15), index tie-break, select-only
  float ob = __shfl_xor(best, 16, 32);
  int   oi = __shfl_xor(bi,   16, 32);
  bool take = (ob < best) | ((ob == best) & (oi < bi));
  best = take ? ob : best;
  bi   = take ? oi : bi;
  if (hl == 0 && row < Nrows) idx[row] = bi;
}

__global__ void zero2_kernel(float* p) { if (threadIdx.x < 2) p[threadIdx.x] = 0.f; }

__global__ void vq_gather_loss_kernel(const float* __restrict__ z,
                                      const float* __restrict__ W,
                                      const int* __restrict__ idx,
                                      float* __restrict__ hq,
                                      float* __restrict__ loss, int Nrows, int D) {
  int r = blockIdx.x * blockDim.x + threadIdx.x;
  if (r >= Nrows) return;
  int j = idx[r];
  const float4* wr = (const float4*)(W + j * D);
  const float4* zr = (const float4*)(z + r * D);
  float4*       hr = (float4*)(hq + r * D);
  float s = 0.f;
  for (int q = 0; q < D / 4; ++q) {
    float4 wv = wr[q], zv = zr[q];
    hr[q] = wv;
    float dx = zv.x - wv.x, dy = zv.y - wv.y, dz = zv.z - wv.z, dw = zv.w - wv.w;
    s += dx * dx + dy * dy + dz * dz + dw * dw;
  }
  float c = s / (float)Nrows;
  atomicAdd(&loss[0], c);   // commit and embed losses are equal in forward
  atomicAdd(&loss[1], c);
}

// ---------------------------------------------------------------------------
extern "C" void kernel_launch(void* const* d_in, const int* in_sizes, int n_in,
                              void* d_out, int out_size, void* d_ws, size_t ws_size,
                              hipStream_t stream) {
  (void)in_sizes; (void)n_in; (void)ws_size;
  const int Bn = 256;
  auto P = [&](int i) { return (const float*)d_in[i]; };
  float* out  = (float*)d_out;
  float* loss = out + (out_size - 2);

  // workspace: 4 ping-pong buffers (each fits every stage: 256*64*256 floats)
  const size_t BIG = (size_t)Bn * 64 * 256;
  float* bufA  = (float*)d_ws;
  float* bufB  = bufA + BIG;
  float* bufC  = bufB + BIG;
  float* bufD  = bufC + BIG;
  float* stats = bufD + BIG;          // 512 f32
  float* scsh  = stats + 512;         // 512 f32
  float* wn    = scsh + 512;          // 1024 f32
  int*   vidx  = (int*)(wn + 1024);   // 65536 i32

  auto conv = [&](const float* x, int Cin, int Lin, const float* w, const float* bias,
                  float* y, int Cout, int Nout, int Lout, int Ksz, int pad,
                  int in_step, int out_step, int out_off, int wmode, int wtap) {
    dim3 grid((unsigned)((Bn * Nout + 31) / 32), (unsigned)((Cout + 15) / 16));
    if (wmode == 0)
      conv1d_wmma_kernel<0><<<grid, 32, 0, stream>>>(x, w, bias, y, Bn, Cin, Lin,
          Cout, Nout, Lout, Ksz, pad, in_step, out_step, out_off, wtap);
    else if (wmode == 1)
      conv1d_wmma_kernel<1><<<grid, 32, 0, stream>>>(x, w, bias, y, Bn, Cin, Lin,
          Cout, Nout, Lout, Ksz, pad, in_step, out_step, out_off, wtap);
    else
      conv1d_wmma_kernel<2><<<grid, 32, 0, stream>>>(x, w, bias, y, Bn, Cin, Lin,
          Cout, Nout, Lout, Ksz, pad, in_step, out_step, out_off, wtap);
  };
  auto bn = [&](const float* y, const float* g, const float* beta,
                const float* addsrc, float* o, bool relu, int C, int L) {
    bn_stats_kernel<<<C, 256, 0, stream>>>(y, stats, Bn, C, L);
    bn_finalize_kernel<<<1, 256, 0, stream>>>(stats, g, beta, scsh, C,
                                              1.f / (float)(Bn * L));
    bn_apply_kernel<<<Bn * C, 128, 0, stream>>>(y, scsh, addsrc, o,
                                                relu ? 1 : 0, C, L);
  };

  const float* X = P(0);

  // ================= encoder =================
  conv(X, 4, 256, P(1), P(2), bufA, 64, 256, 256, 7, 3, 1, 1, 0, 0, 0);
  bn(bufA, P(3), P(4), nullptr, bufB, false, 64, 256);                 // h
  // res1
  conv(bufB, 64, 256, P(5), P(6), bufA, 64, 256, 256, 5, 2, 1, 1, 0, 0, 0);
  bn(bufA, P(7), P(8), nullptr, bufC, true, 64, 256);
  conv(bufC, 64, 256, P(9), P(10), bufA, 64, 256, 256, 5, 2, 1, 1, 0, 0, 0);
  bn(bufA, P(11), P(12), bufB, bufD, false, 64, 256);                  // r1
  // ds1 + ch12 + res2
  conv(bufD, 64, 256, P(91), P(92), bufA, 64, 128, 128, 2, 0, 2, 1, 0, 0, 0);
  conv(bufA, 64, 128, P(83), P(84), bufB, 128, 128, 128, 1, 0, 1, 1, 0, 0, 0);
  conv(bufA, 64, 128, P(13), P(14), bufC, 128, 128, 128, 5, 2, 1, 1, 0, 0, 0);
  bn(bufC, P(15), P(16), nullptr, bufD, true, 128, 128);
  conv(bufD, 128, 128, P(17), P(18), bufC, 128, 128, 128, 5, 2, 1, 1, 0, 0, 0);
  bn(bufC, P(19), P(20), nullptr, bufA, true, 128, 128);
  conv(bufA, 128, 128, P(21), P(22), bufC, 128, 128, 128, 5, 2, 1, 1, 0, 0, 0);
  bn(bufC, P(23), P(24), bufB, bufD, false, 128, 128);                 // r2
  // ds2 + ch23 + res3
  conv(bufD, 128, 128, P(93), P(94), bufA, 128, 64, 64, 2, 0, 2, 1, 0, 0, 0);
  conv(bufA, 128, 64, P(85), P(86), bufB, 256, 64, 64, 1, 0, 1, 1, 0, 0, 0);
  conv(bufA, 128, 64, P(25), P(26), bufC, 256, 64, 64, 3, 1, 1, 1, 0, 0, 0);
  bn(bufC, P(27), P(28), nullptr, bufD, true, 256, 64);
  conv(bufD, 256, 64, P(29), P(30), bufC, 256, 64, 64, 3, 1, 1, 1, 0, 0, 0);
  bn(bufC, P(31), P(32), nullptr, bufA, true, 256, 64);
  conv(bufA, 256, 64, P(33), P(34), bufC, 256, 64, 64, 3, 1, 1, 1, 0, 0, 0);
  bn(bufC, P(35), P(36), bufB, bufD, false, 256, 64);                  // r3
  // ds3 -> z  [B,256,32] == z_e [65536,32]
  conv(bufD, 256, 64, P(95), P(96), bufA, 256, 32, 32, 2, 0, 2, 1, 0, 0, 0);

  // ================= vector quantization =================
  const float* W = P(103);
  wnorm_kernel<<<4, 256, 0, stream>>>(W, wn, 1024, 32);
  vq_argmin_kernel<<<65536 / 16, 32, 0, stream>>>(bufA, W, wn, vidx, 65536, 1024);
  zero2_kernel<<<1, 32, 0, stream>>>(loss);
  vq_gather_loss_kernel<<<65536 / 256, 256, 0, stream>>>(bufA, W, vidx, bufB,
                                                         loss, 65536, 32);  // hq=bufB

  // ================= decoder =================
  // us3 (convT s=2 k=2): two parity GEMMs  [B,256,32] -> [B,256,64]
  conv(bufB, 256, 32, P(97), P(98), bufA, 256, 32, 64, 1, 0, 1, 2, 0, 2, 0);
  conv(bufB, 256, 32, P(97), P(98), bufA, 256, 32, 64, 1, 0, 1, 2, 1, 2, 1);
  // ch32 shortcut + deres3 (convT wmode=1, k5 p2)
  conv(bufA, 256, 64, P(87), P(88), bufB, 128, 64, 64, 1, 0, 1, 1, 0, 0, 0);
  conv(bufA, 256, 64, P(37), P(38), bufC, 128, 64, 64, 5, 2, 1, 1, 0, 1, 0);
  bn(bufC, P(39), P(40), nullptr, bufD, true, 128, 64);
  conv(bufD, 128, 64, P(41), P(42), bufC, 128, 64, 64, 5, 2, 1, 1, 0, 1, 0);
  bn(bufC, P(43), P(44), nullptr, bufA, true, 128, 64);
  conv(bufA, 128, 64, P(45), P(46), bufC, 128, 64, 64, 5, 2, 1, 1, 0, 1, 0);
  bn(bufC, P(47), P(48), nullptr, bufD, true, 128, 64);
  conv(bufD, 128, 64, P(49), P(50), bufC, 128, 64, 64, 5, 2, 1, 1, 0, 1, 0);
  bn(bufC, P(51), P(52), bufB, bufA, false, 128, 64);                  // d3
  // us2  [B,128,64] -> [B,128,128]
  conv(bufA, 128, 64, P(99), P(100), bufB, 128, 64, 128, 1, 0, 1, 2, 0, 2, 0);
  conv(bufA, 128, 64, P(99), P(100), bufB, 128, 64, 128, 1, 0, 1, 2, 1, 2, 1);
  // ch21 shortcut + deres2 (k5 p2)
  conv(bufB, 128, 128, P(89), P(90), bufA, 64, 128, 128, 1, 0, 1, 1, 0, 0, 0);
  conv(bufB, 128, 128, P(53), P(54), bufC, 64, 128, 128, 5, 2, 1, 1, 0, 1, 0);
  bn(bufC, P(55), P(56), nullptr, bufD, true, 64, 128);
  conv(bufD, 64, 128, P(57), P(58), bufC, 64, 128, 128, 5, 2, 1, 1, 0, 1, 0);
  bn(bufC, P(59), P(60), nullptr, bufB, true, 64, 128);
  conv(bufB, 64, 128, P(61), P(62), bufC, 64, 128, 128, 5, 2, 1, 1, 0, 1, 0);
  bn(bufC, P(63), P(64), nullptr, bufD, true, 64, 128);
  conv(bufD, 64, 128, P(65), P(66), bufC, 64, 128, 128, 5, 2, 1, 1, 0, 1, 0);
  bn(bufC, P(67), P(68), bufA, bufB, false, 64, 128);                  // d2
  // us1  [B,64,128] -> [B,64,256]
  conv(bufB, 64, 128, P(101), P(102), bufA, 64, 128, 256, 1, 0, 1, 2, 0, 2, 0);
  conv(bufB, 64, 128, P(101), P(102), bufA, 64, 128, 256, 1, 0, 1, 2, 1, 2, 1);
  // deres1 (k7 p3), shortcut = bufA
  conv(bufA, 64, 256, P(69), P(70), bufC, 64, 256, 256, 7, 3, 1, 1, 0, 1, 0);
  bn(bufC, P(71), P(72), nullptr, bufD, true, 64, 256);
  conv(bufD, 64, 256, P(73), P(74), bufC, 64, 256, 256, 7, 3, 1, 1, 0, 1, 0);
  bn(bufC, P(75), P(76), nullptr, bufB, true, 64, 256);
  conv(bufB, 64, 256, P(77), P(78), bufC, 64, 256, 256, 7, 3, 1, 1, 0, 1, 0);
  bn(bufC, P(79), P(80), bufA, bufD, false, 64, 256);                  // d1
  // deconv (convT 64->4, k7 p3) -> recon directly into d_out
  conv(bufD, 64, 256, P(81), P(82), out, 4, 256, 256, 7, 3, 1, 1, 0, 1, 0);
}
